// Net_84756884619418
// MI455X (gfx1250) — matrix operand order repaired and checked by reading, not compiled, not run on hardware
//
#include <hip/hip_runtime.h>
#include <hip/hip_bf16.h>

#define N_NODES 100000
#define E_EDGES 1600000
#define B_GR    512
#define F_COLS  9
#define V_VOCAB 128
#define H_DIM   256
#define C_CLS   10

typedef __attribute__((ext_vector_type(16))) __bf16    v16bf;
typedef __attribute__((ext_vector_type(8)))  float     v8f;
typedef __attribute__((ext_vector_type(4)))  unsigned  v4u;
typedef __attribute__((ext_vector_type(8)))  int       v8i;
typedef __attribute__((ext_vector_type(4)))  int       v4i;

__device__ __forceinline__ unsigned short f32_to_bf16(float f) {
    unsigned u = __float_as_uint(f);
    u += 0x7FFFu + ((u >> 16) & 1u);          // round to nearest even
    return (unsigned short)(u >> 16);
}
__device__ __forceinline__ float bf16_to_f32(unsigned short h) {
    return __uint_as_float(((unsigned)h) << 16);
}

// ---------------- AtomEncoder: h[n,c] = sum_f emb[f, x[n,f], c]  (bf16 out) ----
__global__ void k_encode(const int* __restrict__ x, const float* __restrict__ emb,
                         unsigned short* __restrict__ hb) {
    const int n = blockIdx.x;
    const int c = threadIdx.x;                // H_DIM threads
    float s = 0.f;
#pragma unroll
    for (int f = 0; f < F_COLS; ++f) {
        const int idx = x[n * F_COLS + f];
        s += emb[((size_t)f * V_VOCAB + idx) * H_DIM + c];
    }
    hb[(size_t)n * H_DIM + c] = f32_to_bf16(s);
}

// ---------------- degrees with self loops ------------------------------------
__global__ void k_deg_init(float* __restrict__ dis) {
    int n = blockIdx.x * 256 + threadIdx.x;
    if (n < N_NODES) dis[n] = 1.0f;           // self loop
}
__global__ void k_deg_edges(const int* __restrict__ ei, float* __restrict__ dis) {
    int e = blockIdx.x * 256 + threadIdx.x;
    if (e < E_EDGES) atomicAdd(&dis[ei[E_EDGES + e]], 1.0f);   // col = edge_index[1]
}
__global__ void k_rsqrt(float* __restrict__ dis) {
    int n = blockIdx.x * 256 + threadIdx.x;
    if (n < N_NODES) dis[n] = rsqrtf(dis[n]);
}
__global__ void k_counts(const int* __restrict__ batch, float* __restrict__ counts) {
    int n = blockIdx.x * 256 + threadIdx.x;
    if (n < N_NODES) atomicAdd(&counts[batch[n]], 1.0f);
}

// ---------------- one-time W convert + transpose: WT[n][k] = bf16(W[k][n]) ----
__global__ void k_wconv(const float* __restrict__ W, unsigned short* __restrict__ WT) {
    const int k = blockIdx.x;                 // 256
    const int n = threadIdx.x;                // 256
    WT[n * H_DIM + k] = f32_to_bf16(W[k * H_DIM + n]);
}

// ---------------- WMMA GEMM: C[N,256](bf16) = A[N,256](bf16) @ W ---------------
// WT is bf16 [n][k] (k contiguous). 8 waves/block, each wave owns 16 rows.
// W chunks (64 cols x 256 k = 32KB) staged into LDS by the Tensor Data Mover
// with pad_enable so the LDS row stride becomes 512+16B = 264 ushorts.
union ABFrag { uint4 q[2]; v16bf v; };

#define LDS_STRIDE 264                         // ushorts per staged column

__device__ __forceinline__ void tdm_load_chunk(const unsigned short* gsrc,
                                               unsigned lds_off) {
    const unsigned long long ga = (unsigned long long)(size_t)gsrc;
    v4u g0;
    g0.x = 0x1u;                                           // count=1, user mode
    g0.y = lds_off;                                        // lds_addr (bytes)
    g0.z = (unsigned)ga;                                   // global_addr[31:0]
    g0.w = (unsigned)((ga >> 32) & 0x1FFFFFFull) | (2u << 30);  // addr[56:32], type=2
    v8i g1;
    g1[0] = (int)((1u << 16)      // data_size = 1 -> 2 bytes
                | (1u << 20)      // pad_enable
                | (6u << 22)      // pad_interval = 6 -> 128 dwords (512B row)
                | (3u << 25));    // pad_amount   = 3 -> 4 dwords (16B)
    g1[1] = (int)(256u << 16);    // tensor_dim0[15:0]=256 (k extent)
    g1[2] = (int)(256u << 16);    // tensor_dim1[15:0]=256 (rows in WT)
    g1[3] = (int)(256u << 16);    // tile_dim0 = 256 (k per column)
    g1[4] = 64;                   // tile_dim1 = 64 columns
    g1[5] = 256;                  // tensor_dim0_stride = 256 elements
    g1[6] = 0;
    g1[7] = 0;
    v4i gz4 = {0, 0, 0, 0};       // 2-D tensor: groups 2/3 unused
    v8i gz8 = {0, 0, 0, 0, 0, 0, 0, 0};
    __builtin_amdgcn_tensor_load_to_lds(g0, g1, gz4, gz4, gz8, 0);
}

__launch_bounds__(256, 2)
__global__ void k_gemm(const unsigned short* __restrict__ A,
                       const unsigned short* __restrict__ WT,
                       unsigned short* __restrict__ Cmat, int nrows) {
    __shared__ unsigned short ldsW[2][64 * LDS_STRIDE];   // double buffer, 2x33KB
    const int t    = threadIdx.x;
    const int wave = t >> 5;
    const int lane = t & 31;
    const int half = lane >> 4;                   // 0: lanes 0-15, 1: lanes 16-31
    const int l16  = lane & 15;
    const int rowBase = blockIdx.x * 128 + wave * 16;

    // Preload A fragments for the 8 k-steps (16x32 bf16 tile per step).
    // ISA layout: lanes 0-15 VGPR0-3 = K0..7, VGPR4-7 = K16..23;
    //             lanes 16-31 get K8..15 / K24..31.
    ABFrag afrag[8];
    {
        int r = rowBase + l16;
        if (r >= nrows) r = nrows - 1;            // clamp: keep EXEC all-1s for WMMA
        const unsigned short* arow = A + (size_t)r * H_DIM;
#pragma unroll
        for (int kk = 0; kk < 8; ++kk) {
            const int kb = kk * 32 + half * 8;
            afrag[kk].q[0] = *(const uint4*)(arow + kb);        // K kb..kb+7
            afrag[kk].q[1] = *(const uint4*)(arow + kb + 16);   // K kb+16..kb+23
        }
    }

    if (wave == 0)                                 // prefetch chunk 0 -> buf 0
        tdm_load_chunk(WT, (unsigned)(size_t)&ldsW[0][0]);

    const bool fullTile = (rowBase + 16) <= nrows; // wave-uniform store fast path

    for (int chunk = 0; chunk < 4; ++chunk) {      // 4 x 64 output columns
        const int buf = chunk & 1;
        __syncthreads();                           // prev compute done on buf^1
        if (wave == 0) {
            if (chunk < 3) {                       // prefetch next chunk
                tdm_load_chunk(WT + (size_t)(chunk + 1) * 64 * H_DIM,
                               (unsigned)(size_t)&ldsW[buf ^ 1][0]);
                __builtin_amdgcn_s_wait_tensorcnt(1);   // current chunk landed
            } else {
                __builtin_amdgcn_s_wait_tensorcnt(0);
            }
        }
        __syncthreads();                           // broadcast chunk ready

        const int colOff = chunk * 64;
#pragma unroll
        for (int ct = 0; ct < 4; ++ct) {
            v8f acc = {};
            // B layout: lanes 0-15 hold K0..7 / K16..23 of each 32-chunk, lanes 16-31 K8..15 / K24..31
            const unsigned short* wcol =
                &ldsW[buf][(ct * 16 + l16) * LDS_STRIDE + half * 8];
#pragma unroll
            for (int kk = 0; kk < 8; ++kk) {
                ABFrag b;
                const unsigned short* wp = wcol + kk * 32;
                b.q[0] = *(const uint4*)(wp);
                b.q[1] = *(const uint4*)(wp + 16);
                acc = __builtin_amdgcn_wmma_f32_16x16x32_bf16(
                        false, afrag[kk].v, false, b.v, (short)0, acc, false, false);
            }
            const int col = colOff + ct * 16 + l16;
            unsigned short* crow = Cmat + (size_t)(rowBase + half * 8) * H_DIM + col;
            if (fullTile) {
#pragma unroll
                for (int v = 0; v < 8; ++v)        // C layout: VGPR v -> M = v + half*8
                    crow[(size_t)v * H_DIM] = f32_to_bf16(acc[v]);
            } else {
#pragma unroll
                for (int v = 0; v < 8; ++v) {
                    const int r = rowBase + half * 8 + v;
                    if (r < nrows)
                        Cmat[(size_t)r * H_DIM + col] = f32_to_bf16(acc[v]);
                }
            }
        }
    }
}

// ---------------- edge scatter: agg[col] += norm * m[row] --------------------
__launch_bounds__(256)
__global__ void k_scatter(const int* __restrict__ ei, const float* __restrict__ dis,
                          const unsigned short* __restrict__ m, float* __restrict__ agg) {
    const long total = (long)E_EDGES + N_NODES;
    const long e = (long)blockIdx.x * 8 + (threadIdx.x >> 5);   // one wave per edge
    if (e >= total) return;
    const int lane = threadIdx.x & 31;
    int row, col;
    if (e < E_EDGES) { row = ei[e]; col = ei[E_EDGES + e]; }
    else             { row = (int)(e - E_EDGES); col = row; }   // self loop
    const float norm = dis[row] * dis[col];
    const unsigned* mrow = (const unsigned*)(m + (size_t)row * H_DIM);
    float* acol = agg + (size_t)col * H_DIM;
#pragma unroll
    for (int i = 0; i < 4; ++i) {
        const int c2 = lane + i * 32;             // uint index -> cols 2*c2, 2*c2+1
        const unsigned mm = mrow[c2];
        atomicAdd(acol + 2 * c2,     norm * bf16_to_f32((unsigned short)(mm & 0xFFFFu)));
        atomicAdd(acol + 2 * c2 + 1, norm * bf16_to_f32((unsigned short)(mm >> 16)));
    }
}

// ---------------- bias + ReLU + bf16 re-emit + pool accumulate + agg reset ----
__global__ void k_finalize(float* __restrict__ agg, const float* __restrict__ bias,
                           const int* __restrict__ batch, unsigned short* __restrict__ hb,
                           float* __restrict__ pooled) {
    const int n = blockIdx.x;
    const int c = threadIdx.x;
    const size_t idx = (size_t)n * H_DIM + c;
    float v = agg[idx] + bias[c];
    v = fmaxf(v, 0.0f);
    agg[idx] = 0.0f;                              // ready for next layer
    hb[idx] = f32_to_bf16(v);                     // next layer input
    atomicAdd(&pooled[(size_t)batch[n] * H_DIM + c], v);
}

// ---------------- classifier --------------------------------------------------
__global__ void k_cls1(const float* __restrict__ pooled, const float* __restrict__ counts,
                       const float* __restrict__ cw1, const float* __restrict__ cb1,
                       float* __restrict__ hdn) {
    const int g = blockIdx.x;                     // B_GR
    const int j = threadIdx.x;                    // 128
    const float inv = 1.0f / counts[g];
    float s = cb1[j];
    for (int k = 0; k < H_DIM; ++k)
        s = fmaf(pooled[g * H_DIM + k] * inv, cw1[k * 128 + j], s);
    hdn[g * 128 + j] = fmaxf(s, 0.0f);
}
__global__ void k_cls2(const float* __restrict__ hdn, const float* __restrict__ cw2,
                       const float* __restrict__ cb2, float* __restrict__ out) {
    const int g = blockIdx.x;
    const int c = threadIdx.x;                    // 32, first 10 active
    if (c < C_CLS) {
        float s = cb2[c];
        for (int j = 0; j < 128; ++j)
            s = fmaf(hdn[g * 128 + j], cw2[j * C_CLS + c], s);
        out[g * C_CLS + c] = s;
    }
}

extern "C" void kernel_launch(void* const* d_in, const int* in_sizes, int n_in,
                              void* d_out, int out_size, void* d_ws, size_t ws_size,
                              hipStream_t stream) {
    const int*   x     = (const int*)d_in[0];
    const int*   ei    = (const int*)d_in[1];
    const int*   batch = (const int*)d_in[2];
    const float* emb   = (const float*)d_in[3];
    const float* W1    = (const float*)d_in[4];
    const float* b1    = (const float*)d_in[5];
    const float* W2    = (const float*)d_in[6];
    const float* b2    = (const float*)d_in[7];
    const float* W3    = (const float*)d_in[8];
    const float* b3    = (const float*)d_in[9];
    const float* cw1   = (const float*)d_in[10];
    const float* cb1   = (const float*)d_in[11];
    const float* cw2   = (const float*)d_in[12];
    const float* cb2   = (const float*)d_in[13];
    float* out = (float*)d_out;
    (void)in_sizes; (void)n_in; (void)ws_size;

    char* ws = (char*)d_ws;
    size_t off = 0;
    auto take = [&](size_t bytes) -> void* {
        void* p = ws + off;
        off = (off + bytes + 255) & ~(size_t)255;
        return p;
    };
    unsigned short* hb   = (unsigned short*)take((size_t)N_NODES * H_DIM * 2); // bf16 feats
    unsigned short* msg  = (unsigned short*)take((size_t)N_NODES * H_DIM * 2); // bf16 feat@W
    float* agg    = (float*)take((size_t)N_NODES * H_DIM * 4);
    float* dis    = (float*)take((size_t)N_NODES * 4);
    float* pooled = (float*)take((size_t)B_GR * H_DIM * 4);
    float* counts = (float*)take((size_t)B_GR * 4);
    float* hdn    = (float*)take((size_t)B_GR * 128 * 4);
    unsigned short* WT = (unsigned short*)take((size_t)H_DIM * H_DIM * 2);     // bf16 W^T

    (void)hipMemsetAsync(agg,    0, (size_t)N_NODES * H_DIM * 4, stream);
    (void)hipMemsetAsync(pooled, 0, (size_t)B_GR * H_DIM * 4, stream);
    (void)hipMemsetAsync(counts, 0, (size_t)B_GR * 4, stream);
    (void)hipMemsetAsync(d_out,  0, (size_t)out_size * 4, stream); // incl. trailing 0 scalar

    k_encode   <<<N_NODES, H_DIM, 0, stream>>>(x, emb, hb);
    k_deg_init <<<(N_NODES + 255) / 256, 256, 0, stream>>>(dis);
    k_deg_edges<<<(E_EDGES + 255) / 256, 256, 0, stream>>>(ei, dis);
    k_rsqrt    <<<(N_NODES + 255) / 256, 256, 0, stream>>>(dis);
    k_counts   <<<(N_NODES + 255) / 256, 256, 0, stream>>>(batch, counts);

    const float* Ws[3] = {W1, W2, W3};
    const float* bs[3] = {b1, b2, b3};
    const unsigned gemmGrid    = (N_NODES + 127) / 128;
    const unsigned scatterGrid = (unsigned)(((long)E_EDGES + N_NODES + 7) / 8);
    for (int l = 0; l < 3; ++l) {
        k_wconv   <<<H_DIM, H_DIM, 0, stream>>>(Ws[l], WT);
        k_gemm    <<<gemmGrid, 256, 0, stream>>>(hb, WT, msg, N_NODES);
        k_scatter <<<scatterGrid, 256, 0, stream>>>(ei, dis, msg, agg);
        k_finalize<<<N_NODES, H_DIM, 0, stream>>>(agg, bs[l], batch, hb, pooled);
    }
    k_cls1<<<B_GR, 128, 0, stream>>>(pooled, counts, cw1, cb1, hdn);
    k_cls2<<<B_GR, 32, 0, stream>>>(hdn, cw2, cb2, out);
}
